// PartMambaDenoiser_29781303230908
// MI455X (gfx1250) — matrix operand order repaired and checked by reading, not compiled, not run on hardware
//
#include <hip/hip_runtime.h>
#include <cstdint>
#include <cstddef>

// ---------------- types ----------------
typedef __bf16 bh;
typedef __attribute__((ext_vector_type(16))) __bf16 v16bf;
typedef __attribute__((ext_vector_type(8)))  float  v8f;

#define B_   256
#define NP_  3
#define LAT_ 512
#define TXT_ 768
#define D_   768
#define L_   5
#define NL_  6
#define DI_  1536
#define DS_  16
#define DTR_ 48

__device__ __forceinline__ float bf2f(bh x) {
  unsigned short u = __builtin_bit_cast(unsigned short, x);
  unsigned v = ((unsigned)u) << 16;
  return __builtin_bit_cast(float, v);
}
__device__ __forceinline__ bh f2bf(float f) {
  unsigned u = __builtin_bit_cast(unsigned, f);
  u += 0x7FFFu + ((u >> 16) & 1u);
  unsigned short h = (unsigned short)(u >> 16);
  return __builtin_bit_cast(bh, h);
}
__device__ __forceinline__ float sigm(float x) { return 1.f / (1.f + __expf(-x)); }

// ---------------- WMMA GEMM: C[M,N] = A[M,K](bf16) * W[N,K]^T(bf16) ----------------
// 128 threads (4 waves). Block tile 128(M) x 64(N), K-step 32.
// LDS double-buffered, filled with global_load_async_to_lds_b128 (ASYNCcnt-tracked
// DMA, no VGPR staging). Each wave: 2 M-subtiles x 4 N-subtiles = 8 wmma / K-step.
// LDS rows padded to 20 dwords -> conflict-free ds_load_b128 fragment reads.
// Requirements: M % 128 == 0, K % 32 == 0, W has >= ceil64(N) rows (zero padded).
// Epilogue: +bias, act (0=none,1=silu,2=softplus), optional f32 out, optional bf16 out.
#define RS_ 5  // LDS row stride in uint4 (16 dwords data + 4 pad)
__global__ __launch_bounds__(128)
void k_gemm(const bh* __restrict__ A, int lda,
            const bh* __restrict__ W, int ldw,
            float* __restrict__ C, int ldc,
            bh* __restrict__ Cb, int ldcb,
            const float* __restrict__ bias,
            int M, int N, int K, int act)
{
  __shared__ uint4 As[2][128 * RS_];
  __shared__ uint4 Bs[2][64 * RS_];

  const int tid  = threadIdx.x;
  const int wave = tid >> 5;     // 0..3 -> 32-row M strip
  const int lane = tid & 31;
  const int g    = lane >> 4;
  const int hl   = lane & 15;
  const int m0   = blockIdx.y * 128;
  const int n0   = blockIdx.x * 64;

  v8f acc[2][4];
#pragma unroll
  for (int s = 0; s < 2; ++s)
#pragma unroll
    for (int t = 0; t < 4; ++t) acc[s][t] = (v8f){0.f,0.f,0.f,0.f,0.f,0.f,0.f,0.f};

  // Per-thread load slice: uint4 column fixed (tid&3), rows stride 32.
  const int rr = tid >> 2;          // 0..31
  const int cc = tid & 3;           // uint4 column within 32-bf16 row chunk
  const bh* apB = A + (size_t)(m0 + rr) * lda + 8 * cc;
  const bh* wpB = W + (size_t)(n0 + rr) * ldw + 8 * cc;

  // async DMA of one 128x32 + 64x32 bf16 tile pair into LDS buffer q
  auto fetch = [&](int q, int k0) {
#pragma unroll
    for (int i = 0; i < 4; ++i) {
      uint32_t lds = (uint32_t)(uintptr_t)&As[q][(rr + i * 32) * RS_ + cc];
      uint64_t ga  = (uint64_t)(uintptr_t)(apB + (size_t)i * 32 * lda + k0);
      asm volatile("global_load_async_to_lds_b128 %0, %1, off"
                   :: "v"(lds), "v"(ga) : "memory");
    }
#pragma unroll
    for (int i = 0; i < 2; ++i) {
      uint32_t lds = (uint32_t)(uintptr_t)&Bs[q][(rr + i * 32) * RS_ + cc];
      uint64_t ga  = (uint64_t)(uintptr_t)(wpB + (size_t)i * 32 * ldw + k0);
      asm volatile("global_load_async_to_lds_b128 %0, %1, off"
                   :: "v"(lds), "v"(ga) : "memory");
    }
  };

  fetch(0, 0);

  const int nk = K >> 5;
  for (int kt = 0; kt < nk; ++kt) {
    const int p = kt & 1;
    // my async DMAs for buffer p have landed:
    asm volatile("s_wait_asynccnt 0x0" ::: "memory");
    // everyone's DMAs visible; previous readers of buffer 1-p are done:
    __syncthreads();
    if (kt + 1 < nk) {
      fetch(1 - p, (kt + 1) << 5);          // overlaps with compute below
      if (kt + 2 < nk) {                     // speculative L2 prefetch
        const int k2 = (kt + 2) << 5;
        __builtin_prefetch(apB + k2, 0, 1);
        __builtin_prefetch(wpB + k2, 0, 1);
      }
    }

    const uint32_t* as = (const uint32_t*)As[p];
    const uint32_t* bs = (const uint32_t*)Bs[p];

    union { v16bf v; uint32_t u[8]; } af[2];
#pragma unroll
    for (int s = 0; s < 2; ++s)
#pragma unroll
      for (int j = 0; j < 8; ++j) {
        int h = j >> 2, q2 = j & 3;
        int kk = 16 * h + 8 * g + 2 * q2;  // even K offset within tile
        af[s].u[j] = as[(wave * 32 + s * 16 + hl) * (RS_ * 4) + (kk >> 1)];
      }
#pragma unroll
    for (int t = 0; t < 4; ++t) {
      union { v16bf v; uint32_t u[8]; } bfr;
#pragma unroll
      for (int j = 0; j < 8; ++j)
        bfr.u[j] = bs[(t * 16 + hl) * (RS_ * 4) + 8 * (j >> 2) + 4 * g + (j & 3)];
      acc[0][t] = __builtin_amdgcn_wmma_f32_16x16x32_bf16(
          false, af[0].v, false, bfr.v, (short)0, acc[0][t], false, false);
      acc[1][t] = __builtin_amdgcn_wmma_f32_16x16x32_bf16(
          false, af[1].v, false, bfr.v, (short)0, acc[1][t], false, false);
    }
  }

  // epilogue (C/D: n = hl, m = e + 8*g per 16x16 tile)
#pragma unroll
  for (int s = 0; s < 2; ++s)
#pragma unroll
    for (int t = 0; t < 4; ++t) {
      int n = n0 + t * 16 + hl;
      if (n >= N) continue;
      float bv = bias ? bias[n] : 0.f;
#pragma unroll
      for (int e = 0; e < 8; ++e) {
        int m = m0 + wave * 32 + s * 16 + e + 8 * g;
        float val = acc[s][t][e] + bv;
        if (act == 1)      val = val * sigm(val);
        else if (act == 2) val = (val > 20.f) ? val : log1pf(__expf(val));
        if (C)  C[(size_t)m * ldc + n]   = val;
        if (Cb) Cb[(size_t)m * ldcb + n] = f2bf(val);
      }
    }
}

// ---------------- aux kernels ----------------
__global__ __launch_bounds__(256)
void k_cvt(const float* __restrict__ s, bh* __restrict__ d, size_t n) {
  size_t i = (size_t)blockIdx.x * 256 + threadIdx.x;
  if (i < n) d[i] = f2bf(s[i]);
}

// convert [N,K] f32 -> [Npad,K] bf16 with zero row padding
__global__ __launch_bounds__(256)
void k_cvt_wpad(const float* __restrict__ s, bh* __restrict__ d,
                int N, int K, int Npad) {
  int i = blockIdx.x * 256 + threadIdx.x;
  if (i >= Npad * K) return;
  int r = i / K;
  d[i] = (r < N) ? f2bf(s[i]) : f2bf(0.f);
}

// convert with column padding: [rows, scols] (stride sstride) f32 -> [rows, dcols] bf16
__global__ __launch_bounds__(256)
void k_cvt_pad(const float* __restrict__ s, int sstride, int scols,
               bh* __restrict__ d, int dcols, int rows) {
  int i = blockIdx.x * 256 + threadIdx.x;
  int tot = rows * dcols;
  if (i >= tot) return;
  int r = i / dcols, c = i - r * dcols;
  float v = (c < scols) ? s[(size_t)r * sstride + c] : 0.f;
  d[i] = f2bf(v);
}

__global__ __launch_bounds__(256)
void k_time_embed(const int* __restrict__ ts, bh* __restrict__ temb) {
  int i = blockIdx.x * 256 + threadIdx.x;
  if (i >= B_ * D_) return;
  int b = i / D_, d = i - b * D_;
  const int half = D_ / 2;
  float t = (float)ts[b];
  int j = (d < half) ? d : d - half;
  float freq = __expf(-logf(10000.f) * (float)j / (float)half);
  float a = t * freq;
  temb[i] = f2bf((d < half) ? cosf(a) : sinf(a));
}

__global__ __launch_bounds__(256)
void k_assemble(const float* __restrict__ zf, const float* __restrict__ tf,
                const float* __restrict__ tif, const float* __restrict__ part_emb,
                float* __restrict__ x, bh* __restrict__ xb) {
  int i = blockIdx.x * 256 + threadIdx.x;
  if (i >= B_ * L_ * D_) return;
  int d = i % D_;
  int l = (i / D_) % L_;
  int b = i / (D_ * L_);
  float v;
  if (l == 0)      v = tif[(size_t)b * D_ + d];
  else if (l == 1) v = tf[(size_t)b * D_ + d];
  else             v = zf[((size_t)b * NP_ + (l - 2)) * D_ + d] + part_emb[(size_t)(l - 2) * D_ + d];
  int j = d >> 1;
  float ang = (float)l * __expf(-(float)(2 * j) * (logf(10000.f) / (float)D_));
  v += (d & 1) ? cosf(ang) : sinf(ang);
  x[i] = v;
  xb[i] = f2bf(v);
}

__global__ __launch_bounds__(256)
void k_copy_tc(const float* __restrict__ x, float* __restrict__ tc) {
  int i = blockIdx.x * 256 + threadIdx.x;
  if (i >= B_ * D_) return;
  int b = i / D_, d = i - b * D_;
  tc[i] = x[((size_t)b * L_ + 1) * D_ + d];
}

__global__ __launch_bounds__(256)
void k_fill_rep(const float* __restrict__ tc, bh* __restrict__ X1) {
  int i = blockIdx.x * 256 + threadIdx.x;
  if (i >= B_ * L_ * D_) return;
  int d = i % D_;
  int row = i / D_;
  int b = row / L_;
  X1[(size_t)row * (2 * D_) + D_ + d] = f2bf(tc[(size_t)b * D_ + d]);
}

__global__ __launch_bounds__(256)
void k_gate(const float* __restrict__ yhat, const float* __restrict__ tc,
            bh* __restrict__ X1) {
  int i = blockIdx.x * 256 + threadIdx.x;
  if (i >= B_ * L_ * D_) return;
  int d = i % D_;
  int row = i / D_;
  int b = row / L_;
  float t = tc[(size_t)b * D_ + d];
  X1[(size_t)row * (2 * D_) + D_ + d] = f2bf(t * sigm(yhat[i]));
}

// causal conv (DC=4) + silu; flip handled by source index remap
__global__ __launch_bounds__(256)
void k_conv(const float* __restrict__ xz, const float* __restrict__ cw,
            const float* __restrict__ cb, float* __restrict__ uc,
            bh* __restrict__ ucb, int flip) {
  int i = blockIdx.x * 256 + threadIdx.x;
  if (i >= B_ * L_ * DI_) return;
  int c = i % DI_;
  int l = (i / DI_) % L_;
  int b = i / (DI_ * L_);
  float acc = cb[c];
#pragma unroll
  for (int k = 0; k < 4; ++k) {
    int ls = l + k - 3;
    if (ls >= 0) {
      int lsrc = flip ? (L_ - 1 - ls) : ls;
      acc += cw[(size_t)c * 4 + k] * xz[((size_t)b * L_ + lsrc) * (2 * DI_) + c];
    }
  }
  acc = acc * sigm(acc);
  uc[i] = acc;
  ucb[i] = f2bf(acc);
}

// selective scan over L=5; writes silu(z)-gated output (bf16) at original positions
__global__ __launch_bounds__(256)
void k_scan(const float* __restrict__ dt, const float* __restrict__ xdbl,
            const float* __restrict__ uc, const float* __restrict__ xz,
            const float* __restrict__ Alog, const float* __restrict__ Dp,
            bh* __restrict__ ybuf, int flip) {
  int b = blockIdx.y;
  int d = blockIdx.x * 256 + threadIdx.x;
  __shared__ float Bs[L_ * DS_], Cs[L_ * DS_];
  if (threadIdx.x < L_ * DS_) {
    int l = threadIdx.x / DS_, s = threadIdx.x % DS_;
    Bs[threadIdx.x] = xdbl[((size_t)b * L_ + l) * 80 + DTR_ + s];
    Cs[threadIdx.x] = xdbl[((size_t)b * L_ + l) * 80 + DTR_ + DS_ + s];
  }
  __syncthreads();
  float Av[DS_], h[DS_];
#pragma unroll
  for (int s = 0; s < DS_; ++s) {
    Av[s] = -__expf(Alog[(size_t)d * DS_ + s]);
    h[s] = 0.f;
  }
  float Dv = Dp[d];
  for (int l = 0; l < L_; ++l) {
    float dtv = dt[((size_t)b * L_ + l) * DI_ + d];
    float uv  = uc[((size_t)b * L_ + l) * DI_ + d];
    float y = 0.f;
#pragma unroll
    for (int s = 0; s < DS_; ++s) {
      float dA = __expf(dtv * Av[s]);
      h[s] = h[s] * dA + dtv * uv * Bs[l * DS_ + s];
      y += h[s] * Cs[l * DS_ + s];
    }
    y += Dv * uv;
    int lo = flip ? (L_ - 1 - l) : l;
    float z = xz[((size_t)b * L_ + lo) * (2 * DI_) + DI_ + d];
    y *= z * sigm(z);
    ybuf[((size_t)b * L_ + lo) * DI_ + d] = f2bf(y);
  }
}

// (optional residual add) + layernorm; zmode=1 extracts x[:,2:,:] rows
__global__ __launch_bounds__(256)
void k_addln(const float* __restrict__ src, const float* __restrict__ res,
             const float* __restrict__ gam, const float* __restrict__ bet,
             float* __restrict__ outf, bh* __restrict__ outb, int zmode, int Dd) {
  int row = blockIdx.x;
  size_t base = (zmode == 1)
      ? ((size_t)(row / NP_) * L_ + 2 + (row % NP_)) * Dd
      : (size_t)row * Dd;
  float loc[4];
  int cnt = 0;
  float s = 0.f, q = 0.f;
  for (int d = threadIdx.x; d < Dd; d += 256) {
    float v = src[base + d];
    if (res) v += res[base + d];
    loc[cnt++] = v;
    s += v; q += v * v;
  }
  __shared__ float s1[256], s2[256];
  s1[threadIdx.x] = s; s2[threadIdx.x] = q;
  __syncthreads();
  for (int off = 128; off > 0; off >>= 1) {
    if (threadIdx.x < off) { s1[threadIdx.x] += s1[threadIdx.x + off]; s2[threadIdx.x] += s2[threadIdx.x + off]; }
    __syncthreads();
  }
  float mean = s1[0] / (float)Dd;
  float var  = s2[0] / (float)Dd - mean * mean;
  float inv  = rsqrtf(var + 1e-5f);
  cnt = 0;
  for (int d = threadIdx.x; d < Dd; d += 256) {
    float v = (loc[cnt++] - mean) * inv * gam[d] + bet[d];
    if (outf) outf[base + d] = v;
    if (outb) outb[((zmode == 1) ? (size_t)row * Dd : base) + d] = f2bf(v);
  }
}

// ---------------- host helpers ----------------
static void run_gemm(const bh* A, int lda, const float* Wf, bh* Wslab,
                     int M, int N, int K, const float* bias,
                     float* C, int ldc, bh* Cb, int ldcb, int act,
                     hipStream_t stream) {
  int Npad = (N + 63) & ~63;
  size_t wn = (size_t)Npad * K;
  k_cvt_wpad<<<dim3((unsigned)((wn + 255) / 256)), dim3(256), 0, stream>>>(Wf, Wslab, N, K, Npad);
  dim3 grid(Npad / 64, M / 128);
  k_gemm<<<grid, dim3(128), 0, stream>>>(A, lda, Wslab, K, C, ldc, Cb, ldcb, bias, M, N, K, act);
}

extern "C" void kernel_launch(void* const* d_in, const int* in_sizes, int n_in,
                              void* d_out, int out_size, void* d_ws, size_t ws_size,
                              hipStream_t stream) {
  (void)in_sizes; (void)n_in; (void)out_size; (void)ws_size;
  const float* z_noisy  = (const float*)d_in[0];
  const float* text_emb = (const float*)d_in[1];
  const float* zin_w  = (const float*)d_in[2];
  const float* zin_b  = (const float*)d_in[3];
  const float* tin_w  = (const float*)d_in[4];
  const float* tin_b  = (const float*)d_in[5];
  const float* time_w = (const float*)d_in[6];
  const float* time_b = (const float*)d_in[7];
  const float* part_emb = (const float*)d_in[8];
  const float* local_w = (const float*)d_in[9];
  const float* local_b = (const float*)d_in[10];
  const float* f_w    = (const float*)d_in[11];
  const float* f_b    = (const float*)d_in[12];
  const float* fuse_w = (const float*)d_in[13];
  const float* fuse_b = (const float*)d_in[14];
  const float* final_w = (const float*)d_in[15];
  const float* final_b = (const float*)d_in[16];
  const float* ln_g = (const float*)d_in[17];
  const float* ln_b = (const float*)d_in[18];
  const float* m_in_w    = (const float*)d_in[19];
  const float* m_conv_w  = (const float*)d_in[20];
  const float* m_conv_b  = (const float*)d_in[21];
  const float* m_xproj_w = (const float*)d_in[22];
  const float* m_dt_w    = (const float*)d_in[23];
  const float* m_dt_b    = (const float*)d_in[24];
  const float* m_Alog    = (const float*)d_in[25];
  const float* m_Dp      = (const float*)d_in[26];
  const float* m_out_w   = (const float*)d_in[27];
  const float* oln_g  = (const float*)d_in[28];
  const float* oln_b  = (const float*)d_in[29];
  const float* zout_w = (const float*)d_in[30];
  const float* zout_b = (const float*)d_in[31];
  const int*   timestep = (const int*)d_in[32];

  char* p = (char*)d_ws;
  auto carve = [&](size_t bytes) -> char* {
    char* r = p;
    p += (bytes + 255) & ~(size_t)255;
    return r;
  };
  const int MR = B_ * L_;  // 1280 token rows
  bh*    Wslab = (bh*)carve((size_t)2 * DI_ * D_ * 2);     // max weight (3072x768)
  float* x     = (float*)carve((size_t)MR * D_ * 4);
  bh*    xb    = (bh*)carve((size_t)MR * D_ * 2);
  float* tc    = (float*)carve((size_t)B_ * D_ * 4);
  bh*    X1    = (bh*)carve((size_t)MR * 2 * D_ * 2);
  bh*    X2    = (bh*)carve((size_t)MR * 2 * D_ * 2);
  float* yhat  = (float*)carve((size_t)MR * D_ * 4);
  float* xt    = (float*)carve((size_t)MR * D_ * 4);
  bh*    xtb   = (bh*)carve((size_t)MR * D_ * 2);
  float* xz    = (float*)carve((size_t)MR * 2 * DI_ * 4);
  float* uc    = (float*)carve((size_t)MR * DI_ * 4);
  bh*    ucb   = (bh*)carve((size_t)MR * DI_ * 2);
  float* xdbl  = (float*)carve((size_t)MR * 80 * 4);
  bh*    dtA   = (bh*)carve((size_t)MR * 64 * 2);
  float* dtf   = (float*)carve((size_t)MR * DI_ * 4);
  bh*    ybuf  = (bh*)carve((size_t)MR * DI_ * 2);
  float* xo    = (float*)carve((size_t)MR * D_ * 4);
  float* zf    = (float*)carve((size_t)B_ * NP_ * D_ * 4);
  float* tfb   = (float*)carve((size_t)B_ * D_ * 4);
  float* tif   = (float*)carve((size_t)B_ * D_ * 4);
  bh*    temb  = (bh*)carve((size_t)B_ * D_ * 2);
  bh*    znb   = (bh*)carve((size_t)B_ * NP_ * LAT_ * 2);
  bh*    teb   = (bh*)carve((size_t)B_ * TXT_ * 2);
  bh*    zb    = (bh*)carve((size_t)B_ * NP_ * D_ * 2);

  // ---- embedding ----
  {
    size_t n = (size_t)B_ * NP_ * LAT_;
    k_cvt<<<dim3((unsigned)((n + 255) / 256)), dim3(256), 0, stream>>>(z_noisy, znb, n);
    n = (size_t)B_ * TXT_;
    k_cvt<<<dim3((unsigned)((n + 255) / 256)), dim3(256), 0, stream>>>(text_emb, teb, n);
  }
  k_time_embed<<<dim3((B_ * D_ + 255) / 256), dim3(256), 0, stream>>>(timestep, temb);
  run_gemm(znb, LAT_, zin_w, Wslab, B_ * NP_, D_, LAT_, zin_b, zf, D_, nullptr, 0, 0, stream);
  run_gemm(teb, TXT_, tin_w, Wslab, B_, D_, TXT_, tin_b, tfb, D_, nullptr, 0, 0, stream);
  run_gemm(temb, D_, time_w, Wslab, B_, D_, D_, time_b, tif, D_, nullptr, 0, 0, stream);
  k_assemble<<<dim3((MR * D_ + 255) / 256), dim3(256), 0, stream>>>(zf, tfb, tif, part_emb, x, xb);
  k_copy_tc<<<dim3((B_ * D_ + 255) / 256), dim3(256), 0, stream>>>(x, tc);

  const int ELW = (MR * D_ + 255) / 256;
  // ---- layers ----
  for (int i = 0; i < NL_; ++i) {
    // xl = silu(x @ local_w^T + b) -> bf16 left half of X1
    run_gemm(xb, D_, local_w + (size_t)i * D_ * D_, Wslab, MR, D_, D_,
             local_b + (size_t)i * D_, nullptr, 0, X1, 2 * D_, 1, stream);
    k_fill_rep<<<dim3(ELW), dim3(256), 0, stream>>>(tc, X1);
    // yhat = [xl, y_rep] @ f_w^T + f_b
    run_gemm(X1, 2 * D_, f_w + (size_t)i * D_ * 2 * D_, Wslab, MR, D_, 2 * D_,
             f_b + (size_t)i * D_, yhat, D_, nullptr, 0, 0, stream);
    k_gate<<<dim3(ELW), dim3(256), 0, stream>>>(yhat, tc, X1);
    // xt = [xl, y_gate] @ fuse_w^T + fuse_b
    run_gemm(X1, 2 * D_, fuse_w + (size_t)i * D_ * 2 * D_, Wslab, MR, D_, 2 * D_,
             fuse_b + (size_t)i * D_, xt, D_, xtb, D_, 0, stream);

    for (int dir = 0; dir < 2; ++dir) {
      size_t wo = (size_t)(i * 2 + dir);
      // xz = xt @ in_w^T  (position independent -> flip folded downstream)
      run_gemm(xtb, D_, m_in_w + wo * 2 * DI_ * D_, Wslab, MR, 2 * DI_, D_,
               nullptr, xz, 2 * DI_, nullptr, 0, 0, stream);
      k_conv<<<dim3((MR * DI_ + 255) / 256), dim3(256), 0, stream>>>(
          xz, m_conv_w + wo * DI_ * 4, m_conv_b + wo * DI_, uc, ucb, dir);
      // xdbl = uc @ xproj_w^T   (N=80 -> padded W rows to 128)
      run_gemm(ucb, DI_, m_xproj_w + wo * (DTR_ + 2 * DS_) * DI_, Wslab, MR,
               DTR_ + 2 * DS_, DI_, nullptr, xdbl, 80, nullptr, 0, 0, stream);
      // dt = softplus(xdbl[:, :48] @ dt_w^T + dt_b), K padded 48->64
      k_cvt_pad<<<dim3((MR * 64 + 255) / 256), dim3(256), 0, stream>>>(xdbl, 80, DTR_, dtA, 64, MR);
      k_cvt_pad<<<dim3((DI_ * 64 + 255) / 256), dim3(256), 0, stream>>>(
          m_dt_w + wo * DI_ * DTR_, DTR_, DTR_, Wslab, 64, DI_);
      k_gemm<<<dim3(DI_ / 64, MR / 128), dim3(128), 0, stream>>>(
          dtA, 64, Wslab, 64, dtf, DI_, nullptr, 0, m_dt_b + wo * DI_, MR, DI_, 64, 2);
      // selective scan + Dp*uc + silu(z) gate -> ybuf (bf16, original positions)
      k_scan<<<dim3(DI_ / 256, B_), dim3(256), 0, stream>>>(
          dtf, xdbl, uc, xz, m_Alog + wo * DI_ * DS_, m_Dp + wo * DI_, ybuf, dir);
      // xf / xb = ybuf @ out_w^T -> halves of X2 (bf16)
      run_gemm(ybuf, DI_, m_out_w + wo * D_ * DI_, Wslab, MR, D_, DI_,
               nullptr, nullptr, 0, X2 + dir * D_, 2 * D_, 0, stream);
    }
    // xo = [xf, xb] @ final_w^T + final_b
    run_gemm(X2, 2 * D_, final_w + (size_t)i * D_ * 2 * D_, Wslab, MR, D_, 2 * D_,
             final_b + (size_t)i * D_, xo, D_, nullptr, 0, 0, stream);
    // x = layernorm(xo + x)
    k_addln<<<dim3(MR), dim3(256), 0, stream>>>(xo, x, ln_g + (size_t)i * D_,
                                                ln_b + (size_t)i * D_, x, xb, 0, D_);
  }

  // ---- output head ----
  k_addln<<<dim3(B_ * NP_), dim3(256), 0, stream>>>(x, nullptr, oln_g, oln_b,
                                                    nullptr, zb, 1, D_);
  run_gemm(zb, D_, zout_w, Wslab, B_ * NP_, LAT_, D_, zout_b,
           (float*)d_out, LAT_, nullptr, 0, 0, stream);
}
